// MambaMixer_81664508166612
// MI455X (gfx1250) — matrix operand order repaired and checked by reading, not compiled, not run on hardware
//
#include <hip/hip_runtime.h>

#define BSZ 2
#define LSEQ 2048
#define HDIM 2048
#define EDIM 4096
#define NSSM 16
#define KCONV 4
#define RRANK 128
#define NPROJ 160   // R + 2N

typedef __attribute__((ext_vector_type(8)))  float   v8f;
typedef __attribute__((ext_vector_type(16))) __bf16  v16bf;
typedef __attribute__((ext_vector_type(8)))  __bf16  v8bf;

// async-to-LDS builtin operand types: pointer to int4 (vector_size 16),
// global (AS1) source and LDS (AS3) destination.
typedef int v4i_t __attribute__((vector_size(16)));
typedef __attribute__((address_space(1))) v4i_t gv4i;
typedef __attribute__((address_space(3))) v4i_t lv4i;

#if defined(__has_builtin)
#  if __has_builtin(__builtin_amdgcn_global_load_async_to_lds_b128)
#    define ASYNC_LDS 1
#  endif
#endif
#ifndef ASYNC_LDS
#  define ASYNC_LDS 0
#endif

__device__ __forceinline__ void wait_asynccnt0() {
#if defined(__has_builtin) && __has_builtin(__builtin_amdgcn_s_wait_asynccnt)
    __builtin_amdgcn_s_wait_asynccnt(0);
#else
    asm volatile("s_wait_asynccnt 0x0" ::: "memory");
#endif
}

// ---------------------------------------------------------------------------
// f32 -> bf16 conversion
// ---------------------------------------------------------------------------
__global__ void cvt_f32_bf16(const float* __restrict__ in, __bf16* __restrict__ out, int n) {
    int i = blockIdx.x * blockDim.x + threadIdx.x;
    if (i < n) out[i] = (__bf16)in[i];
}

// ---------------------------------------------------------------------------
// WMMA fragment loaders (bf16 16x16x32).
// A layout: lane r<16 holds K={0..7,16..23}; lanes 16-31 hold K={8..15,24..31}.
// B layout: lane n<16 holds K={0..7,16..23} of column n; lanes 16-31 hold K={8..15,24..31}.
// ---------------------------------------------------------------------------
__device__ __forceinline__ v16bf load_a_frag(const __bf16* __restrict__ A, int ld,
                                             int m0, int k0, int lane) {
    int r    = lane & 15;
    int koff = (lane >> 4) << 3;
    const __bf16* p = A + (size_t)(m0 + r) * ld + k0 + koff;
    v8bf lo = *(const v8bf*)(p);
    v8bf hi = *(const v8bf*)(p + 16);
    return __builtin_shufflevector(lo, hi, 0,1,2,3,4,5,6,7,8,9,10,11,12,13,14,15);
}

__device__ __forceinline__ v16bf load_a_frag_lds(const __bf16* As, int row0, int lane) {
    int r    = lane & 15;
    int koff = (lane >> 4) << 3;
    const __bf16* p = As + (row0 + r) * 32 + koff;
    v8bf lo = *(const v8bf*)(p);
    v8bf hi = *(const v8bf*)(p + 16);
    return __builtin_shufflevector(lo, hi, 0,1,2,3,4,5,6,7,8,9,10,11,12,13,14,15);
}

__device__ __forceinline__ v16bf load_b_frag(const __bf16* __restrict__ W, int ld,
                                             int n0, int k0, int lane) {
    int n    = n0 + (lane & 15);
    int koff = (lane >> 4) << 3;
    const __bf16* p = W + (size_t)n * ld + k0 + koff;
    v8bf lo = *(const v8bf*)(p);
    v8bf hi = *(const v8bf*)(p + 16);
    return __builtin_shufflevector(lo, hi, 0,1,2,3,4,5,6,7,8,9,10,11,12,13,14,15);
}

// ---------------------------------------------------------------------------
// Blocked bf16 WMMA GEMM: C[m,n] = sum_k A[m,k]*W[n,k], fp32 accumulate.
// 128 threads = 4 waves (2x2); each wave owns a 64x64 tile (4x4 WMMA tiles,
// 16 accumulators) -> block tile 128x128. A K-slab (128x32 bf16, 8KB) is
// staged in LDS per K-step (async-to-LDS when available); B weight fragments
// stream from global with one-K-step-ahead prefetch.
// MODE 0: row-major f32 store (ldo)
// MODE 1: in_proj split: n<E -> u [B,E,L]; n>=E -> gate [B*L,E]
// MODE 2: +bias[n] + softplus, row-major store
// ---------------------------------------------------------------------------
template <int MODE>
__global__ void __launch_bounds__(128)
gemm_blocked_wmma(const __bf16* __restrict__ A, const __bf16* __restrict__ W,
                  float* __restrict__ out0, float* __restrict__ out1,
                  const float* __restrict__ bias, int K, int ldo) {
    __shared__ __align__(64) __bf16 As[128][32];

    const int tid  = threadIdx.x;
    const int lane = tid & 31;
    const int wave = tid >> 5;
    const int wm   = wave >> 1;
    const int wn   = wave & 1;
    const int mblk = blockIdx.y * 128;
    const int n0   = blockIdx.x * 128 + wn * 64;
    const int m0   = wm * 64;                     // within block tile

    v8f acc[4][4];
    const v8f vzero = {0.f, 0.f, 0.f, 0.f, 0.f, 0.f, 0.f, 0.f};
#pragma unroll
    for (int mt = 0; mt < 4; ++mt)
#pragma unroll
        for (int nt = 0; nt < 4; ++nt) acc[mt][nt] = vzero;

    const __bf16* arow = A + (size_t)(mblk + tid) * K;   // one A row per thread

    for (int k = 0; k < K; k += 32) {
        __syncthreads();                                  // protect prior reads
#if ASYNC_LDS
        {
            gv4i* g = (gv4i*)(const void*)(arow + k);
            lv4i* l = (lv4i*)(void*)&As[tid][0];
            __builtin_amdgcn_global_load_async_to_lds_b128(g,     l,     0, 0);
            __builtin_amdgcn_global_load_async_to_lds_b128(g + 1, l + 1, 0, 0);
            __builtin_amdgcn_global_load_async_to_lds_b128(g + 2, l + 2, 0, 0);
            __builtin_amdgcn_global_load_async_to_lds_b128(g + 3, l + 3, 0, 0);
            wait_asynccnt0();
        }
#else
        *(v16bf*)&As[tid][0]  = *(const v16bf*)(arow + k);
        *(v16bf*)&As[tid][16] = *(const v16bf*)(arow + k + 16);
#endif
        __syncthreads();

        // prefetch next K-step of the B (weight) stream: one line per tile row
        if (k + 32 < K) {
#pragma unroll
            for (int nt = 0; nt < 4; ++nt)
                __builtin_prefetch(
                    (const void*)(W + (size_t)(n0 + 16 * nt + (lane & 15)) * K + k + 32),
                    0, 1);
        }

        v16bf af[4], bf[4];
#pragma unroll
        for (int mt = 0; mt < 4; ++mt)
            af[mt] = load_a_frag_lds(&As[0][0], m0 + 16 * mt, lane);
#pragma unroll
        for (int nt = 0; nt < 4; ++nt)
            bf[nt] = load_b_frag(W, K, n0 + 16 * nt, k, lane);

#pragma unroll
        for (int mt = 0; mt < 4; ++mt)
#pragma unroll
            for (int nt = 0; nt < 4; ++nt)
                acc[mt][nt] = __builtin_amdgcn_wmma_f32_16x16x32_bf16(
                    false, af[mt], false, bf[nt], (short)0, acc[mt][nt], false, false);
    }

    const int cn  = lane & 15;
    const int cmb = (lane >> 4) * 8;
#pragma unroll
    for (int mt = 0; mt < 4; ++mt) {
#pragma unroll
        for (int nt = 0; nt < 4; ++nt) {
            int n = n0 + 16 * nt + cn;
#pragma unroll
            for (int v = 0; v < 8; ++v) {
                int   m   = mblk + m0 + 16 * mt + cmb + v;
                float val = acc[mt][nt][v];
                if (MODE == 0) {
                    out0[(size_t)m * ldo + n] = val;
                } else if (MODE == 1) {
                    int b = m >> 11;
                    int l = m & (LSEQ - 1);
                    if (n < EDIM)
                        out0[((size_t)b * EDIM + n) * LSEQ + l] = val;
                    else
                        out1[(size_t)m * EDIM + (n - EDIM)] = val;
                } else {
                    val += bias[n];
                    val = (val > 20.f) ? val : logf(1.f + __expf(val));
                    out0[(size_t)m * ldo + n] = val;
                }
            }
        }
    }
}

// ---------------------------------------------------------------------------
// Small one-wave GEMM (for x_proj, N=160): 16x16 tile per wave.
// ---------------------------------------------------------------------------
__global__ void gemm_small_wmma(const __bf16* __restrict__ A, const __bf16* __restrict__ W,
                                float* __restrict__ out0, int K, int ldo) {
    const int lane = threadIdx.x;          // blockDim.x == 32
    const int m0   = blockIdx.y * 16;
    const int n0   = blockIdx.x * 16;
    v8f acc = {0.f, 0.f, 0.f, 0.f, 0.f, 0.f, 0.f, 0.f};
    for (int k = 0; k < K; k += 32) {
        v16bf af = load_a_frag(A, K, m0, k, lane);
        v16bf bf = load_b_frag(W, K, n0, k, lane);
        acc = __builtin_amdgcn_wmma_f32_16x16x32_bf16(
            false, af, false, bf, (short)0, acc, false, false);
    }
    const int cn  = lane & 15;
    const int cmb = (lane >> 4) * 8;
#pragma unroll
    for (int v = 0; v < 8; ++v)
        out0[(size_t)(m0 + cmb + v) * ldo + n0 + cn] = acc[v];
}

// ---------------------------------------------------------------------------
// Causal depthwise conv (K=4) + SiLU. u [B,E,L] -> x f32 [B*L,E] + bf16 [B*L,E].
// ---------------------------------------------------------------------------
__global__ void conv_silu_kernel(const float* __restrict__ U, const float* __restrict__ cw,
                                 const float* __restrict__ cb, float* __restrict__ X,
                                 __bf16* __restrict__ Xbf) {
    size_t idx = (size_t)blockIdx.x * blockDim.x + threadIdx.x;   // B*L*E = 2^24
    int e = (int)(idx & (EDIM - 1));
    int l = (int)((idx >> 12) & (LSEQ - 1));
    int b = (int)(idx >> 23);
    float acc = cb[e];
    const float* up = U + ((size_t)b * EDIM + e) * LSEQ;
#pragma unroll
    for (int k = 0; k < KCONV; ++k) {
        int ll = l - (KCONV - 1) + k;
        if (ll >= 0) acc += up[ll] * cw[e * KCONV + k];
    }
    float sv = acc / (1.f + __expf(-acc));
    X[idx]   = sv;
    Xbf[idx] = (__bf16)sv;
}

// ---------------------------------------------------------------------------
// dt_low slice -> bf16
// ---------------------------------------------------------------------------
__global__ void dtlow_cvt(const float* __restrict__ XDBL, __bf16* __restrict__ out) {
    int i = blockIdx.x * blockDim.x + threadIdx.x;
    int m = i >> 7, c = i & (RRANK - 1);
    out[i] = (__bf16)XDBL[(size_t)m * NPROJ + c];
}

// ---------------------------------------------------------------------------
// Selective scan (fp32), thread per (b,e), N=16 state in registers. B/C step
// vectors staged in LDS. Fused: y = (y + x*D) * silu(gate) -> bf16 [B*L,E].
// ---------------------------------------------------------------------------
__global__ void scan_kernel(const float* __restrict__ DT, const float* __restrict__ X,
                            const float* __restrict__ XDBL, const float* __restrict__ GATE,
                            const float* __restrict__ A_log, const float* __restrict__ Dw,
                            __bf16* __restrict__ YG) {
    int idx = blockIdx.x * blockDim.x + threadIdx.x;  // B*E
    int e = idx & (EDIM - 1);
    int b = idx >> 12;

    float a[NSSM], s[NSSM];
#pragma unroll
    for (int n = 0; n < NSSM; ++n) {
        a[n] = -__expf(A_log[e * NSSM + n]);
        s[n] = 0.f;
    }
    const float De = Dw[e];

    __shared__ float sBC[2 * NSSM];
    for (int l = 0; l < LSEQ; ++l) {
        size_t row = (size_t)b * LSEQ + l;
        __syncthreads();
        if (threadIdx.x < 2 * NSSM)
            sBC[threadIdx.x] = XDBL[row * NPROJ + RRANK + threadIdx.x];
        __syncthreads();

        float dt  = DT[row * EDIM + e];
        float xv  = X[row * EDIM + e];
        float dtu = dt * xv;
        float y   = 0.f;
#pragma unroll
        for (int n = 0; n < NSSM; ++n) {
            float dA = __expf(dt * a[n]);
            s[n] = s[n] * dA + dtu * sBC[n];
            y += s[n] * sBC[NSSM + n];
        }
        float g  = GATE[row * EDIM + e];
        float yt = y + xv * De;
        float yg = yt * (g / (1.f + __expf(-g)));
        YG[row * EDIM + e] = (__bf16)yg;
    }
}

// ---------------------------------------------------------------------------
// Host side
// ---------------------------------------------------------------------------
extern "C" void kernel_launch(void* const* d_in, const int* in_sizes, int n_in,
                              void* d_out, int out_size, void* d_ws, size_t ws_size,
                              hipStream_t stream) {
    const float* hs   = (const float*)d_in[0];
    const float* w1   = (const float*)d_in[1];
    const float* cw   = (const float*)d_in[2];
    const float* cb   = (const float*)d_in[3];
    const float* xpw  = (const float*)d_in[4];
    const float* dtw  = (const float*)d_in[5];
    const float* dtb  = (const float*)d_in[6];
    const float* alog = (const float*)d_in[7];
    const float* dsk  = (const float*)d_in[8];
    const float* ow   = (const float*)d_in[9];
    float* out = (float*)d_out;

    const int M = BSZ * LSEQ;   // 4096
    char* p = (char*)d_ws;
    auto alloc = [&](size_t bytes) -> void* {
        void* r = (void*)p;
        p += (bytes + 255) & ~(size_t)255;
        return r;
    };
    __bf16* HSbf  = (__bf16*)alloc((size_t)M * HDIM * 2);
    __bf16* W1bf  = (__bf16*)alloc((size_t)2 * EDIM * HDIM * 2);
    float*  U     = (float*) alloc((size_t)BSZ * EDIM * LSEQ * 4);
    float*  GATE  = (float*) alloc((size_t)M * EDIM * 4);
    float*  X     = (float*) alloc((size_t)M * EDIM * 4);
    __bf16* XBF   = (__bf16*)alloc((size_t)M * EDIM * 2);
    __bf16* XPWbf = (__bf16*)alloc((size_t)NPROJ * EDIM * 2);
    float*  XDBL  = (float*) alloc((size_t)M * NPROJ * 4);
    __bf16* DTLOW = (__bf16*)alloc((size_t)M * RRANK * 2);
    __bf16* DTWbf = (__bf16*)alloc((size_t)EDIM * RRANK * 2);
    float*  DT    = (float*) alloc((size_t)M * EDIM * 4);
    __bf16* YGbf  = (__bf16*)alloc((size_t)M * EDIM * 2);
    __bf16* OWbf  = (__bf16*)alloc((size_t)HDIM * EDIM * 2);
    (void)ws_size; (void)n_in; (void)in_sizes; (void)out_size;

    // 1) bf16 conversions
    {
        int n;
        n = M * HDIM;          cvt_f32_bf16<<<(n + 255) / 256, 256, 0, stream>>>(hs,  HSbf,  n);
        n = 2 * EDIM * HDIM;   cvt_f32_bf16<<<(n + 255) / 256, 256, 0, stream>>>(w1,  W1bf,  n);
        n = NPROJ * EDIM;      cvt_f32_bf16<<<(n + 255) / 256, 256, 0, stream>>>(xpw, XPWbf, n);
        n = EDIM * RRANK;      cvt_f32_bf16<<<(n + 255) / 256, 256, 0, stream>>>(dtw, DTWbf, n);
        n = HDIM * EDIM;       cvt_f32_bf16<<<(n + 255) / 256, 256, 0, stream>>>(ow,  OWbf,  n);
    }

    // 2) in_proj: [4096,2048] x [8192,2048]^T, split store u/gate
    gemm_blocked_wmma<1><<<dim3((2 * EDIM) / 128, M / 128), 128, 0, stream>>>(
        HSbf, W1bf, U, GATE, nullptr, HDIM, 0);

    // 3) causal depthwise conv + silu
    conv_silu_kernel<<<(BSZ * EDIM * LSEQ) / 256, 256, 0, stream>>>(U, cw, cb, X, XBF);

    // 4) x_proj: [4096,4096] x [160,4096]^T -> x_dbl
    gemm_small_wmma<<<dim3(NPROJ / 16, M / 16), 32, 0, stream>>>(
        XBF, XPWbf, XDBL, EDIM, NPROJ);

    // 5) dt_low slice -> bf16
    dtlow_cvt<<<(M * RRANK) / 256, 256, 0, stream>>>(XDBL, DTLOW);

    // 6) dt_proj + bias + softplus: [4096,128] x [4096,128]^T -> dt
    gemm_blocked_wmma<2><<<dim3(EDIM / 128, M / 128), 128, 0, stream>>>(
        DTLOW, DTWbf, DT, nullptr, dtb, RRANK, EDIM);

    // 7) selective scan (fp32) + fused D-skip + silu gating -> bf16
    scan_kernel<<<(BSZ * EDIM) / 256, 256, 0, stream>>>(DT, X, XDBL, GATE, alog, dsk, YGbf);

    // 8) out_proj: [4096,4096] x [2048,4096]^T -> d_out
    gemm_blocked_wmma<0><<<dim3(HDIM / 128, M / 128), 128, 0, stream>>>(
        YGbf, OWbf, out, nullptr, nullptr, EDIM, HDIM);
}
